// AttnDecoderRNN_23862838296991
// MI455X (gfx1250) — compile-verified, matrix operand order checked by reading, not run on hardware
//
#include <hip/hip_runtime.h>
#include <hip/hip_bf16.h>
#include <math.h>

// ---------------------------------------------------------------------------
// Problem constants (match reference)
// ---------------------------------------------------------------------------
#define Bb 64
#define Ss 512
#define Vv 50257
#define Ee 1024
#define Hh 1024
#define Cc 2048          // 2*H
#define Ll 512           // (H+1)/2
#define DLl 1024         // 2*L
#define XW 3072          // E + C

#define NT 4             // N-subtiles per wave -> block tile 64(M) x 64(N)

typedef __attribute__((ext_vector_type(16))) __bf16 v16bf;
typedef __attribute__((ext_vector_type(8)))  float  v8f;

// ---------------------------------------------------------------------------
// gfx1250 async global->LDS copy (ASYNCcnt-tracked). The builtin's prototype
// (from the compiler diagnostic) takes:
//   arg0: int __vector_size__(16) * in AS1 (global source)
//   arg1: LDS destination pointer (AS3)
//   arg2/arg3: immediate offset / cache policy
// ---------------------------------------------------------------------------
#if defined(__has_builtin)
# if __has_builtin(__builtin_amdgcn_global_load_async_to_lds_b128)
#  define HAVE_ASYNC_LDS 1
# endif
#endif

#ifdef HAVE_ASYNC_LDS
typedef int v4i_ __attribute__((vector_size(16)));
typedef __attribute__((address_space(1))) v4i_ AS1_v4i;
typedef __attribute__((address_space(3))) v4i_ AS3_v4i;
#endif

__device__ __forceinline__ void copy16_to_lds(const float* __restrict__ g,
                                              float* __restrict__ l)
{
#ifdef HAVE_ASYNC_LDS
    #pragma unroll
    for (int j = 0; j < 4; ++j)
        __builtin_amdgcn_global_load_async_to_lds_b128(
            (AS1_v4i*)(g + 4 * j),
            (AS3_v4i*)(l + 4 * j),
            0, 0);
#else
    #pragma unroll
    for (int j = 0; j < 4; ++j)
        reinterpret_cast<float4*>(l)[j] = reinterpret_cast<const float4*>(g)[j];
#endif
}

__device__ __forceinline__ void wait_async_copies()
{
#ifdef HAVE_ASYNC_LDS
# if __has_builtin(__builtin_amdgcn_s_wait_asynccnt)
    __builtin_amdgcn_s_wait_asynccnt(0);
# else
    asm volatile("s_wait_asynccnt 0x0" ::: "memory");
# endif
#endif
}

__device__ __forceinline__ v16bf frag_from_lds(const float* __restrict__ r)
{
    // r points at row_base + 8*half; elements e<8 -> r[e], e>=8 -> r[16+(e-8)]
    float4 a0 = reinterpret_cast<const float4*>(r)[0];
    float4 a1 = reinterpret_cast<const float4*>(r)[1];
    float4 a2 = reinterpret_cast<const float4*>(r + 16)[0];
    float4 a3 = reinterpret_cast<const float4*>(r + 16)[1];
    v16bf f;
    f[0]=(__bf16)a0.x; f[1]=(__bf16)a0.y; f[2]=(__bf16)a0.z; f[3]=(__bf16)a0.w;
    f[4]=(__bf16)a1.x; f[5]=(__bf16)a1.y; f[6]=(__bf16)a1.z; f[7]=(__bf16)a1.w;
    f[8]=(__bf16)a2.x; f[9]=(__bf16)a2.y; f[10]=(__bf16)a2.z; f[11]=(__bf16)a2.w;
    f[12]=(__bf16)a3.x; f[13]=(__bf16)a3.y; f[14]=(__bf16)a3.z; f[15]=(__bf16)a3.w;
    return f;
}

// ---------------------------------------------------------------------------
// WMMA GEMM:  C[64,N] = A[64,K](row-major, lda) * Bw[N,K]^T (+bias) (+acc)
// block = 128 threads = 4 waves; wave w owns output rows [16w,16w+16).
// Block tile 64(M) x 64(N): each wave computes NT=4 16x16 subtiles, reusing
// its A fragment.  A and B tiles staged through LDS (async copies).
// grid.x = ceil(N/64).  bf16 inputs, f32 accumulate.
// Fragment layouts per ISA 7.12.2:
//   A/B element e -> K = (e&7) + 8*half + (e>=8 ? 16 : 0),  half = lane>>4
//   D VGPR r -> row m0 + r + 8*half, col n0 + (lane&15)
// ---------------------------------------------------------------------------
__global__ __launch_bounds__(128)
void gemm_wmma_kernel(const float* __restrict__ A, int lda,
                      const float* __restrict__ Bw,
                      const float* __restrict__ bias,
                      float* __restrict__ C,
                      int N, int K, int acc_flag)
{
    __shared__ __align__(16) float As[64 * 32];   // 64 rows x 32 K   (8 KB)
    __shared__ __align__(16) float Bs[64 * 32];   // 64 cols x 32 K   (8 KB)

    const int tid  = threadIdx.x;
    const int lane = tid & 31;
    const int wave = tid >> 5;
    const int half = lane >> 4;
    const int l16  = lane & 15;
    const int m0   = wave * 16;                 // M = 64 fixed (4 waves)
    const int nblk = blockIdx.x * 64;

    // staging map: thread t stages row ra = t>>1, float chunk ka = 16*(t&1)
    const int ra = tid >> 1;
    const int ka = (tid & 1) * 16;
    const int nb = nblk + ra;                   // weight row staged by this thread
    const bool nb_ok = (nb < N);

    v8f acc[NT] = {};

    for (int k0 = 0; k0 < K; k0 += 32) {
        __syncthreads();                        // previous iter done reading LDS

        // ---- stage A tile (always in range: M == 64) ----
        copy16_to_lds(A + (size_t)ra * lda + k0 + ka, &As[ra * 32 + ka]);

        // ---- stage B tile (tail-guarded along N) ----
        if (nb_ok) {
            copy16_to_lds(Bw + (size_t)nb * K + k0 + ka, &Bs[ra * 32 + ka]);
        } else {
            float4 z = {0.f, 0.f, 0.f, 0.f};
            #pragma unroll
            for (int j = 0; j < 4; ++j)
                reinterpret_cast<float4*>(&Bs[ra * 32 + ka])[j] = z;
        }

        wait_async_copies();                    // own wave's async ops in LDS
        __syncthreads();                        // whole tile visible

        // ---- A fragment (reused across NT subtiles) ----
        const v16bf af = frag_from_lds(&As[(m0 + l16) * 32 + 8 * half]);

        #pragma unroll
        for (int t = 0; t < NT; ++t) {
            const v16bf bfv = frag_from_lds(&Bs[(t * 16 + l16) * 32 + 8 * half]);
            // (neg_a, A, neg_b, B, c_mod, C, reuse_a, reuse_b)
            acc[t] = __builtin_amdgcn_wmma_f32_16x16x32_bf16(
                false, af, false, bfv, (short)0, acc[t], false, false);
        }
    }

    #pragma unroll
    for (int t = 0; t < NT; ++t) {
        const int n = nblk + t * 16 + l16;
        if (n >= N) continue;
        const float bv = bias ? bias[n] : 0.0f;
        #pragma unroll
        for (int r = 0; r < 8; ++r) {
            const int mm = m0 + r + 8 * half;
            float v = acc[t][r] + bv;
            size_t idx = (size_t)mm * N + n;
            if (acc_flag) v += C[idx];
            C[idx] = v;
        }
    }
}

// ---------------------------------------------------------------------------
// e-part of x = [emb[prev_word]; context]
// ---------------------------------------------------------------------------
__global__ void embed_copy_kernel(const int* __restrict__ prev,
                                  const float* __restrict__ emb,
                                  float* __restrict__ x)
{
    const int b = blockIdx.x;
    const int j = threadIdx.x * 4;                    // 256 threads * 4 = 1024 = E
    const float4 v = reinterpret_cast<const float4*>(emb + (size_t)prev[b] * Ee + j)[0];
    reinterpret_cast<float4*>(x + (size_t)b * XW + j)[0] = v;
}

// ---------------------------------------------------------------------------
// logits[b,s] = sum_h tanh(a[b,h] + ctx[b,s,h]) * W_post[h] + b_post
// one block per (s,b); streams attn_contxts once (134 MB).
// ---------------------------------------------------------------------------
__global__ __launch_bounds__(256)
void attn_logits_kernel(const float* __restrict__ a,
                        const float* __restrict__ ctx,
                        const float* __restrict__ Wp,
                        const float* __restrict__ bp,
                        float* __restrict__ logits)
{
    const int s = blockIdx.x, b = blockIdx.y, tid = threadIdx.x;
    const float* ar = a   + (size_t)b * Hh;
    const float* cr = ctx + ((size_t)b * Ss + s) * Hh;
    float p = 0.0f;
    for (int h = tid; h < Hh; h += 256)
        p += tanhf(ar[h] + cr[h]) * Wp[h];
    __shared__ float red[256];
    red[tid] = p; __syncthreads();
    for (int st = 128; st > 0; st >>= 1) {
        if (tid < st) red[tid] += red[tid + st];
        __syncthreads();
    }
    if (tid == 0) logits[(size_t)b * Ss + s] = red[0] + bp[0];
}

// ---------------------------------------------------------------------------
// softmax over S (one block of 512 threads per batch row)
// ---------------------------------------------------------------------------
__global__ __launch_bounds__(512)
void attn_softmax_kernel(const float* __restrict__ logits,
                         float* __restrict__ attn_w)
{
    const int b = blockIdx.x, tid = threadIdx.x;
    __shared__ float red[512];
    const float v = logits[(size_t)b * Ss + tid];
    red[tid] = v; __syncthreads();
    for (int st = 256; st > 0; st >>= 1) {
        if (tid < st) red[tid] = fmaxf(red[tid], red[tid + st]);
        __syncthreads();
    }
    const float mx = red[0]; __syncthreads();
    const float e = __expf(v - mx);
    red[tid] = e; __syncthreads();
    for (int st = 256; st > 0; st >>= 1) {
        if (tid < st) red[tid] += red[tid + st];
        __syncthreads();
    }
    attn_w[(size_t)b * Ss + tid] = e / red[0];
}

// ---------------------------------------------------------------------------
// context[b,c] = sum_s enc[b,s,c] * w[b,s]; written to x[b, E + c].
// Streams encoder_outputs once (268 MB) — dominant HBM cost; prefetch ahead.
// ---------------------------------------------------------------------------
__global__ __launch_bounds__(256)
void context_kernel(const float* __restrict__ enc,
                    const float* __restrict__ w,
                    float* __restrict__ x)
{
    const int b = blockIdx.y;
    const int c = blockIdx.x * 256 + threadIdx.x;
    const float* er = enc + ((size_t)b * Ss) * Cc + c;
    const float* wr = w + (size_t)b * Ss;
    float acc = 0.0f;
    for (int s = 0; s < Ss; ++s) {
        if (s + 8 < Ss)
            __builtin_prefetch(er + (size_t)(s + 8) * Cc, 0, 1);
        acc += er[(size_t)s * Cc] * wr[s];
    }
    x[(size_t)b * XW + Ee + c] = acc;
}

// ---------------------------------------------------------------------------
// LSTM pointwise cell
// ---------------------------------------------------------------------------
__device__ __forceinline__ float sigmoidf(float z) { return 1.0f / (1.0f + __expf(-z)); }

__global__ __launch_bounds__(256)
void lstm_cell_kernel(const float* __restrict__ gates,
                      const float* __restrict__ c_old,
                      float* __restrict__ h_new,
                      float* __restrict__ c_new)
{
    const int idx = blockIdx.x * 256 + threadIdx.x;        // B*H total
    const int b = idx >> 10, h = idx & (Hh - 1);
    const float* g = gates + (size_t)b * (4 * Hh);
    const float gi = g[h], gf = g[Hh + h], gg = g[2 * Hh + h], go = g[3 * Hh + h];
    const float cn = sigmoidf(gf) * c_old[idx] + sigmoidf(gi) * tanhf(gg);
    c_new[idx] = cn;
    h_new[idx] = sigmoidf(go) * tanhf(cn);
}

// ---------------------------------------------------------------------------
// MaxPool1d(2,2):  pooled[b,l] = max(pre[b,2l], pre[b,2l+1])
// ---------------------------------------------------------------------------
__global__ __launch_bounds__(256)
void maxpool_kernel(const float* __restrict__ pre, float* __restrict__ pooled)
{
    const int idx = blockIdx.x * 256 + threadIdx.x;        // B*L total
    const int b = idx >> 9, l = idx & (Ll - 1);
    const float* p = pre + (size_t)b * DLl + 2 * l;
    pooled[idx] = fmaxf(p[0], p[1]);
}

// ---------------------------------------------------------------------------
// In-place log-softmax over V per batch row
// ---------------------------------------------------------------------------
__global__ __launch_bounds__(256)
void logsoftmax_kernel(float* __restrict__ wp)
{
    const int b = blockIdx.x, tid = threadIdx.x;
    float* row = wp + (size_t)b * Vv;
    __shared__ float red[256];
    float mx = -INFINITY;
    for (int i = tid; i < Vv; i += 256) mx = fmaxf(mx, row[i]);
    red[tid] = mx; __syncthreads();
    for (int st = 128; st > 0; st >>= 1) {
        if (tid < st) red[tid] = fmaxf(red[tid], red[tid + st]);
        __syncthreads();
    }
    mx = red[0]; __syncthreads();
    float sum = 0.0f;
    for (int i = tid; i < Vv; i += 256) sum += __expf(row[i] - mx);
    red[tid] = sum; __syncthreads();
    for (int st = 128; st > 0; st >>= 1) {
        if (tid < st) red[tid] += red[tid + st];
        __syncthreads();
    }
    const float lse = mx + __logf(red[0]);
    __syncthreads();
    for (int i = tid; i < Vv; i += 256) row[i] -= lse;
}

// ---------------------------------------------------------------------------
// Host-side orchestration
// ---------------------------------------------------------------------------
extern "C" void kernel_launch(void* const* d_in, const int* in_sizes, int n_in,
                              void* d_out, int out_size, void* d_ws, size_t ws_size,
                              hipStream_t stream)
{
    (void)in_sizes; (void)n_in; (void)out_size; (void)ws_size;

    const int*   prev_word   = (const int*)  d_in[0];
    const float* hidden_s    = (const float*)d_in[1];
    const float* hidden_c    = (const float*)d_in[2];
    const float* encoder_out = (const float*)d_in[3];
    const float* attn_ctx    = (const float*)d_in[4];
    const float* emb         = (const float*)d_in[5];
    const float* W_attn_h    = (const float*)d_in[6];
    const float* attn_bias   = (const float*)d_in[7];
    const float* W_post      = (const float*)d_in[8];
    const float* b_post      = (const float*)d_in[9];
    const float* W_ih        = (const float*)d_in[10];
    const float* W_hh        = (const float*)d_in[11];
    const float* b_lstm      = (const float*)d_in[12];
    const float* W_oh        = (const float*)d_in[13];
    const float* W_oe        = (const float*)d_in[14];
    const float* W_oc        = (const float*)d_in[15];
    const float* b_first     = (const float*)d_in[16];
    const float* W_out       = (const float*)d_in[17];
    const float* b_out       = (const float*)d_in[18];

    // Output layout: word_probs[B,V] | h_new[B,H] | c_new[B,H] | attn_w[B,S,1]
    float* out_wp   = (float*)d_out;
    float* out_h    = out_wp + (size_t)Bb * Vv;
    float* out_c    = out_h  + (size_t)Bb * Hh;
    float* out_attn = out_c  + (size_t)Bb * Hh;

    // Workspace layout (floats)
    float* ws        = (float*)d_ws;
    float* ws_a      = ws;                              // [B,H]
    float* ws_logits = ws_a      + (size_t)Bb * Hh;     // [B,S]
    float* ws_x      = ws_logits + (size_t)Bb * Ss;     // [B,XW] = [e ; context]
    float* ws_gates  = ws_x      + (size_t)Bb * XW;     // [B,4H]
    float* ws_pre    = ws_gates  + (size_t)Bb * 4 * Hh; // [B,DL]
    float* ws_pooled = ws_pre    + (size_t)Bb * DLl;    // [B,L]

    // 1) a = hidden_s @ W_attn_h^T + attn_bias                    [B,H]
    gemm_wmma_kernel<<<(Hh + 63) / 64, 128, 0, stream>>>(
        hidden_s, Hh, W_attn_h, attn_bias, ws_a, Hh, Hh, 0);

    // 2) logits[b,s] = tanh(a + attn_ctx) . W_post + b_post       [B,S]
    attn_logits_kernel<<<dim3(Ss, Bb), 256, 0, stream>>>(
        ws_a, attn_ctx, W_post, b_post, ws_logits);

    // 3) attn_w = softmax_S(logits)  -> directly into d_out region
    attn_softmax_kernel<<<Bb, 512, 0, stream>>>(ws_logits, out_attn);

    // 4) x = [ emb[prev_word] ; context ]
    embed_copy_kernel<<<Bb, 256, 0, stream>>>(prev_word, emb, ws_x);
    context_kernel<<<dim3(Cc / 256, Bb), 256, 0, stream>>>(
        encoder_out, out_attn, ws_x);

    // 5) gates = x @ W_ih^T + b_lstm  ;  gates += hidden_s @ W_hh^T
    gemm_wmma_kernel<<<(4 * Hh + 63) / 64, 128, 0, stream>>>(
        ws_x, XW, W_ih, b_lstm, ws_gates, 4 * Hh, XW, 0);
    gemm_wmma_kernel<<<(4 * Hh + 63) / 64, 128, 0, stream>>>(
        hidden_s, Hh, W_hh, nullptr, ws_gates, 4 * Hh, Hh, 1);

    // 6) LSTM cell -> h_new, c_new (into d_out)
    lstm_cell_kernel<<<(Bb * Hh) / 256, 256, 0, stream>>>(
        ws_gates, hidden_c, out_h, out_c);

    // 7) pre = h_new@W_oh^T + e@W_oe^T + context@W_oc^T + b_first  [B,DL]
    gemm_wmma_kernel<<<(DLl + 63) / 64, 128, 0, stream>>>(
        out_h, Hh, W_oh, b_first, ws_pre, DLl, Hh, 0);
    gemm_wmma_kernel<<<(DLl + 63) / 64, 128, 0, stream>>>(
        ws_x, XW, W_oe, nullptr, ws_pre, DLl, Ee, 1);          // e part of x
    gemm_wmma_kernel<<<(DLl + 63) / 64, 128, 0, stream>>>(
        ws_x + Ee, XW, W_oc, nullptr, ws_pre, DLl, Cc, 1);     // context part

    // 8) pooled = maxpool2(pre)                                    [B,L]
    maxpool_kernel<<<(Bb * Ll) / 256, 256, 0, stream>>>(ws_pre, ws_pooled);

    // 9) word logits = pooled @ W_out^T + b_out  (N=50257 tail-guarded)
    gemm_wmma_kernel<<<(Vv + 63) / 64, 128, 0, stream>>>(
        ws_pooled, Ll, W_out, b_out, out_wp, Vv, Ll, 0);

    // 10) in-place log-softmax over V
    logsoftmax_kernel<<<Bb, 256, 0, stream>>>(out_wp);
}
